// ResidualUnit_15264313770267
// MI455X (gfx1250) — matrix operand order repaired and checked by reading, not compiled, not run on hardware
//
#include <hip/hip_runtime.h>
#include <hip/hip_bf16.h>

// ---------------------------------------------------------------------------
// ConvLSTM residual unit for MI455X (gfx1250), wave32 + WMMA f32_16x16x32_f16.
// Pipeline: ConvLSTM(3->32) -> maxpool(2x2) -> ConvLSTM(32->32) -> upsample
//           -> conv3x3(32->3)+relu -> residual add.
// All convs are implicit GEMMs: M = 16 pixels (row strip), N = 16 out-channels
// per wave (8 waves cover 128 z-channels), K = 32 channels per 3x3 tap.
// ---------------------------------------------------------------------------

typedef __attribute__((ext_vector_type(16))) _Float16 v16h;
typedef __attribute__((ext_vector_type(8)))  _Float16 v8h;
typedef __attribute__((ext_vector_type(8)))  float    v8f;

static constexpr int kB  = 4;
static constexpr int kT  = 16;
static constexpr int kH1 = 128;
static constexpr int kW1 = 128;
static constexpr int kF  = 32;
static constexpr int kH2 = 64;
static constexpr int kW2 = 64;

__device__ __forceinline__ v16h cat16(v8h a, v8h b) {
  return __builtin_shufflevector(a, b, 0,1,2,3,4,5,6,7,8,9,10,11,12,13,14,15);
}
__device__ __forceinline__ v16h ldg16h(const _Float16* __restrict__ p) {
  // 32 contiguous bytes -> two b128 loads
  return cat16(*(const v8h*)p, *(const v8h*)(p + 8));
}
__device__ __forceinline__ v8f wmma_f16(v16h a, v16h b, v8f c) {
  return __builtin_amdgcn_wmma_f32_16x16x32_f16(false, a, false, b, (short)0, c,
                                                false, false);
}
__device__ __forceinline__ float hsig(float v) {
  return fminf(fmaxf(0.2f * v + 0.5f, 0.0f), 1.0f);
}
// A-operand logical K for element e, lane-half hi (ISA 16-bit A 16x32 layout)
__device__ __forceinline__ int kA(int e, int hi) {
  return (e < 8) ? (8 * hi + e) : (16 + 8 * hi + (e - 8));
}

// ---------------------------------------------------------------------------
// Weight packing into WMMA B-operand layout: dst[tap][nt][lane][e] where the
// B layout is K = kA(e, hi) (same order as A), N = lane&15.
// src is [taps][32][coutN] (for W1: taps=1, K space = 27 used of 32).
// ---------------------------------------------------------------------------
__global__ void pack_b_kernel(const float* __restrict__ src,
                              _Float16* __restrict__ dst,
                              int ntiles, int coutN, int kmax, int total) {
  int idx = blockIdx.x * 256 + threadIdx.x;
  if (idx >= total) return;
  int e    = idx & 15;
  int lane = (idx >> 4) & 31;
  int rest = idx >> 9;
  int nt   = rest % ntiles;
  int tap  = rest / ntiles;
  int hi   = lane >> 4, n16 = lane & 15;
  int k    = kA(e, hi);
  int cout = nt * 16 + n16;
  float v = 0.0f;
  if (k < kmax && cout < coutN) v = src[((size_t)tap * 32 + k) * coutN + cout];
  dst[idx] = (_Float16)v;
}

// ---------------------------------------------------------------------------
// Layer-1 ConvLSTM step. Grid: (8 col-tiles, 128 rows, 4 batch), block 256.
// z = conv3x3(x_t, W1) + conv3x3(h_prev, U1) + b1  (128 channels, WMMA)
// gates -> c,h update. h ping-pong (halo race), c in-place (pixel-owned).
// ---------------------------------------------------------------------------
__global__ __launch_bounds__(256)
void lstm1_step_kernel(const float* __restrict__ x, int t,
                       const _Float16* __restrict__ W1pk,  // [1][8][32][16]
                       const _Float16* __restrict__ U1pk,  // [9][8][32][16]
                       const float* __restrict__ b1,
                       const _Float16* __restrict__ hin,   // [4][128][128][32]
                       _Float16* __restrict__ hout,
                       float* __restrict__ c1) {
  const int x0 = blockIdx.x * 16;
  const int y  = blockIdx.y;
  const int b  = blockIdx.z;
  const int tid  = threadIdx.x;
  const int lane = tid & 31;
  const int wave = tid >> 5;

  __shared__ __align__(16) _Float16 ldsH[3 * 18 * 32];  // h halo tile
  __shared__ _Float16 ldsX[3 * 18 * 4];                 // x halo tile (Cin=3+pad)
  __shared__ float    ldsZ[16 * 128];                   // z exchange

  // stage x tile (f32 -> f16), zero pad outside the image
  for (int i = tid; i < 3 * 18; i += 256) {
    int r = i / 18, cc = i % 18;
    int yy = y + r - 1, xx = x0 + cc - 1;
    float v0 = 0.f, v1 = 0.f, v2 = 0.f;
    if ((unsigned)yy < (unsigned)kH1 && (unsigned)xx < (unsigned)kW1) {
      const float* p = x + (((size_t)(b * kT + t) * kH1 + yy) * kW1 + xx) * 3;
      v0 = p[0]; v1 = p[1]; v2 = p[2];
    }
    _Float16* q = ldsX + i * 4;
    q[0] = (_Float16)v0; q[1] = (_Float16)v1;
    q[2] = (_Float16)v2; q[3] = (_Float16)0.f;
  }
  // stage h tile: 54 pixels * 4 x b128 chunks
  for (int i = tid; i < 216; i += 256) {
    int p = i >> 2, ck = i & 3;
    int r = p / 18, cc = p % 18;
    int yy = y + r - 1, xx = x0 + cc - 1;
    int4 val = make_int4(0, 0, 0, 0);
    if ((unsigned)yy < (unsigned)kH1 && (unsigned)xx < (unsigned)kW1) {
      const int4* s =
          (const int4*)(hin + (((size_t)b * kH1 + yy) * kW1 + xx) * kF);
      val = s[ck];
    }
    ((int4*)(ldsH + p * 32))[ck] = val;
  }
  __syncthreads();

  const int nt = wave, hi = lane >> 4, n16 = lane & 15, m = lane & 15;

  float bval = b1[nt * 16 + n16];
  v8f acc;
#pragma unroll
  for (int v = 0; v < 8; ++v) acc[v] = bval;

  // input conv: K space = 9 taps * Cin(3) = 27, one WMMA (K padded to 32)
  v16h ax;
#pragma unroll
  for (int e = 0; e < 16; ++e) {
    int k = kA(e, hi);
    _Float16 val = (_Float16)0.f;
    if (k < 27) {
      int tap = k / 3, cin = k - tap * 3;
      int dy = tap / 3, dx = tap % 3;
      val = ldsX[((dy * 18) + (m + dx)) * 4 + cin];
    }
    ax[e] = val;
  }
  v16h bx = ldg16h(W1pk + ((size_t)nt * 32 + lane) * 16);
  acc = wmma_f16(ax, bx, acc);

  // recurrent conv: 9 taps, K=32 each
#pragma unroll
  for (int tap = 0; tap < 9; ++tap) {
    int dy = tap / 3, dx = tap % 3;
    const _Float16* hp = ldsH + ((dy * 18) + (m + dx)) * 32 + 8 * hi;
    v16h ah = cat16(*(const v8h*)hp, *(const v8h*)(hp + 16));
    v16h bh = ldg16h(U1pk + (((size_t)tap * 8 + nt) * 32 + lane) * 16);
    acc = wmma_f16(ah, bh, acc);
  }

  // D layout: lane half hi, VGPR v -> pixel M = v + 8*hi, channel N = n16
#pragma unroll
  for (int v = 0; v < 8; ++v)
    ldsZ[(v + 8 * hi) * 128 + nt * 16 + n16] = acc[v];
  __syncthreads();

  // gate math: 16 px * 32 features = 512 items
  for (int i = tid; i < 512; i += 256) {
    int px = i >> 5, fc = i & 31;
    float zi = ldsZ[px * 128 + fc];
    float zf = ldsZ[px * 128 + 32 + fc];
    float zg = ldsZ[px * 128 + 64 + fc];
    float zo = ldsZ[px * 128 + 96 + fc];
    size_t gidx = (((size_t)b * kH1 + y) * kW1 + (x0 + px)) * kF + fc;
    float cp = c1[gidx];
    float cn = hsig(zf) * cp + hsig(zi) * fmaxf(zg, 0.f);
    float hn = hsig(zo) * fmaxf(cn, 0.f);
    c1[gidx]   = cn;
    hout[gidx] = (_Float16)hn;
  }
}

// MaxPool 2x2 over h1 -> pooled[t]
__global__ void pool_step_kernel(const _Float16* __restrict__ h,
                                 _Float16* __restrict__ out) {
  int i = blockIdx.x * 256 + threadIdx.x;
  if (i >= kB * kH2 * kW2 * kF) return;
  int fc = i & 31;
  int xx = (i >> 5) & 63;
  int yy = (i >> 11) & 63;
  int b  = i >> 17;
  const _Float16* base = h + ((size_t)b * kH1) * kW1 * kF + fc;
  auto at = [&](int r, int c) {
    return (float)base[((size_t)r * kW1 + c) * kF];
  };
  float v = fmaxf(fmaxf(at(2 * yy, 2 * xx), at(2 * yy, 2 * xx + 1)),
                  fmaxf(at(2 * yy + 1, 2 * xx), at(2 * yy + 1, 2 * xx + 1)));
  out[i] = (_Float16)v;
}

// ---------------------------------------------------------------------------
// Layer-2 ConvLSTM step (64x64, Cin=32 for both convs -> 18 WMMAs/wave).
// Grid: (4 col-tiles, 64 rows, 4 batch), block 256.
// ---------------------------------------------------------------------------
__global__ __launch_bounds__(256)
void lstm2_step_kernel(const _Float16* __restrict__ xin,   // pooled[t]
                       const _Float16* __restrict__ Wpk,   // [9][8][32][16]
                       const _Float16* __restrict__ Upk,   // [9][8][32][16]
                       const float* __restrict__ b2,
                       const _Float16* __restrict__ hin,   // [4][64][64][32]
                       _Float16* __restrict__ hout,
                       float* __restrict__ c2,
                       _Float16* __restrict__ hseq) {      // h2_seq[t]
  const int x0 = blockIdx.x * 16;
  const int y  = blockIdx.y;
  const int b  = blockIdx.z;
  const int tid  = threadIdx.x;
  const int lane = tid & 31;
  const int wave = tid >> 5;

  __shared__ __align__(16) _Float16 ldsXI[3 * 18 * 32];
  __shared__ __align__(16) _Float16 ldsH[3 * 18 * 32];
  __shared__ float ldsZ[16 * 128];

  for (int i = tid; i < 432; i += 256) {  // 216 chunks * 2 tiles
    int which = i >= 216;
    int j = which ? (i - 216) : i;
    int p = j >> 2, ck = j & 3;
    int r = p / 18, cc = p % 18;
    int yy = y + r - 1, xx = x0 + cc - 1;
    int4 val = make_int4(0, 0, 0, 0);
    if ((unsigned)yy < (unsigned)kH2 && (unsigned)xx < (unsigned)kW2) {
      const _Float16* src = which ? hin : xin;
      val = ((const int4*)(src + (((size_t)b * kH2 + yy) * kW2 + xx) * kF))[ck];
    }
    _Float16* dst = which ? ldsH : ldsXI;
    ((int4*)(dst + p * 32))[ck] = val;
  }
  __syncthreads();

  const int nt = wave, hi = lane >> 4, n16 = lane & 15, m = lane & 15;

  float bval = b2[nt * 16 + n16];
  v8f acc;
#pragma unroll
  for (int v = 0; v < 8; ++v) acc[v] = bval;

#pragma unroll
  for (int tap = 0; tap < 9; ++tap) {
    int dy = tap / 3, dx = tap % 3;
    const _Float16* xp = ldsXI + ((dy * 18) + (m + dx)) * 32 + 8 * hi;
    v16h axv = cat16(*(const v8h*)xp, *(const v8h*)(xp + 16));
    v16h bxv = ldg16h(Wpk + (((size_t)tap * 8 + nt) * 32 + lane) * 16);
    acc = wmma_f16(axv, bxv, acc);
  }
#pragma unroll
  for (int tap = 0; tap < 9; ++tap) {
    int dy = tap / 3, dx = tap % 3;
    const _Float16* hp = ldsH + ((dy * 18) + (m + dx)) * 32 + 8 * hi;
    v16h ah = cat16(*(const v8h*)hp, *(const v8h*)(hp + 16));
    v16h bh = ldg16h(Upk + (((size_t)tap * 8 + nt) * 32 + lane) * 16);
    acc = wmma_f16(ah, bh, acc);
  }

#pragma unroll
  for (int v = 0; v < 8; ++v)
    ldsZ[(v + 8 * hi) * 128 + nt * 16 + n16] = acc[v];
  __syncthreads();

  for (int i = tid; i < 512; i += 256) {
    int px = i >> 5, fc = i & 31;
    float zi = ldsZ[px * 128 + fc];
    float zf = ldsZ[px * 128 + 32 + fc];
    float zg = ldsZ[px * 128 + 64 + fc];
    float zo = ldsZ[px * 128 + 96 + fc];
    size_t gidx = (((size_t)b * kH2 + y) * kW2 + (x0 + px)) * kF + fc;
    float cp = c2[gidx];
    float cn = hsig(zf) * cp + hsig(zi) * fmaxf(zg, 0.f);
    float hn = hsig(zo) * fmaxf(cn, 0.f);
    c2[gidx]   = cn;
    _Float16 hh = (_Float16)hn;
    hout[gidx] = hh;
    hseq[gidx] = hh;
  }
}

// ---------------------------------------------------------------------------
// Final: nearest-upsample(h2) -> conv3x3(32->3, padded to N=16) -> relu,
// out + residual. Grid: (128 rows, 64 frames), block 256 (8 waves = 8 strips).
// ---------------------------------------------------------------------------
__global__ __launch_bounds__(256)
void final_conv_kernel(const _Float16* __restrict__ h2seq,  // [16][4][64][64][32]
                       const _Float16* __restrict__ Wcpk,   // [9][32][16]
                       const float* __restrict__ bc,
                       const float* __restrict__ x,
                       float* __restrict__ out,
                       float* __restrict__ res) {
  const int y = blockIdx.x;
  const int n = blockIdx.y;          // frame = b*T + t
  const int b = n >> 4, t = n & 15;
  const int tid  = threadIdx.x;
  const int lane = tid & 31;
  const int wave = tid >> 5;
  const int x0 = wave * 16;
  const int hi = lane >> 4, n16 = lane & 15, m = lane & 15;

  const _Float16* hb = h2seq + (((size_t)t * kB + b) * kH2 * kW2) * kF;

  float bval = (n16 < 3) ? bc[n16] : 0.f;
  v8f acc;
#pragma unroll
  for (int v = 0; v < 8; ++v) acc[v] = bval;

#pragma unroll
  for (int tap = 0; tap < 9; ++tap) {
    int dy = tap / 3 - 1, dx = tap % 3 - 1;
    int yy = y + dy, xx = x0 + m + dx;
    v16h a = {};
    if ((unsigned)yy < (unsigned)kH1 && (unsigned)xx < (unsigned)kW1) {
      const _Float16* p =
          hb + (((size_t)(yy >> 1) * kW2) + (xx >> 1)) * kF + 8 * hi;
      a = cat16(*(const v8h*)p, *(const v8h*)(p + 16));
    }
    v16h bb = ldg16h(Wcpk + ((size_t)tap * 32 + lane) * 16);
    acc = wmma_f16(a, bb, acc);
  }

  if (n16 < 3) {
#pragma unroll
    for (int v = 0; v < 8; ++v) {
      int px = x0 + v + 8 * hi;
      size_t oidx = (((size_t)n * kH1 + y) * kW1 + px) * 3 + n16;
      float val = fmaxf(acc[v], 0.f);
      out[oidx] = val;
      res[oidx] = val + x[oidx];
    }
  }
}

// ---------------------------------------------------------------------------
extern "C" void kernel_launch(void* const* d_in, const int* in_sizes, int n_in,
                              void* d_out, int out_size, void* d_ws,
                              size_t ws_size, hipStream_t stream) {
  (void)in_sizes; (void)n_in; (void)out_size; (void)ws_size;
  const float* x  = (const float*)d_in[0];
  const float* W1 = (const float*)d_in[1];
  const float* U1 = (const float*)d_in[2];
  const float* b1 = (const float*)d_in[3];
  const float* W2 = (const float*)d_in[4];
  const float* U2 = (const float*)d_in[5];
  const float* b2 = (const float*)d_in[6];
  const float* Wc = (const float*)d_in[7];
  const float* bc = (const float*)d_in[8];

  float* out = (float*)d_out;                          // [4,16,128,128,3]
  float* res = out + (size_t)kB * kT * kH1 * kW1 * 3;  // residual

  // workspace carve (~50 MB, 256B aligned)
  char* w = (char*)d_ws;
  auto carve = [&](size_t bytes) {
    char* p = w;
    w += (bytes + 255) & ~(size_t)255;
    return p;
  };
  const size_t h1Elems = (size_t)kB * kH1 * kW1 * kF;  // 2,097,152
  const size_t h2Elems = (size_t)kB * kH2 * kW2 * kF;  //   524,288
  _Float16* W1pk = (_Float16*)carve(1 * 8 * 32 * 16 * 2);
  _Float16* U1pk = (_Float16*)carve(9 * 8 * 32 * 16 * 2);
  _Float16* W2pk = (_Float16*)carve(9 * 8 * 32 * 16 * 2);
  _Float16* U2pk = (_Float16*)carve(9 * 8 * 32 * 16 * 2);
  _Float16* Wcpk = (_Float16*)carve(9 * 1 * 32 * 16 * 2);
  _Float16* h1a  = (_Float16*)carve(h1Elems * 2);
  _Float16* h1b  = (_Float16*)carve(h1Elems * 2);
  float*    c1   = (float*)carve(h1Elems * 4);
  _Float16* hp   = (_Float16*)carve((size_t)kT * h2Elems * 2);
  _Float16* h2a  = (_Float16*)carve(h2Elems * 2);
  _Float16* h2b  = (_Float16*)carve(h2Elems * 2);
  float*    c2   = (float*)carve(h2Elems * 4);
  _Float16* h2s  = (_Float16*)carve((size_t)kT * h2Elems * 2);

  // pack weights into WMMA B layout (every call; deterministic, cheap)
  {
    int tot;
    tot = 1 * 8 * 512;
    pack_b_kernel<<<(tot + 255) / 256, 256, 0, stream>>>(W1, W1pk, 8, 128, 27, tot);
    tot = 9 * 8 * 512;
    pack_b_kernel<<<(tot + 255) / 256, 256, 0, stream>>>(U1, U1pk, 8, 128, 32, tot);
    pack_b_kernel<<<(tot + 255) / 256, 256, 0, stream>>>(W2, W2pk, 8, 128, 32, tot);
    pack_b_kernel<<<(tot + 255) / 256, 256, 0, stream>>>(U2, U2pk, 8, 128, 32, tot);
    tot = 9 * 1 * 512;
    pack_b_kernel<<<(tot + 255) / 256, 256, 0, stream>>>(Wc, Wcpk, 1, 3, 32, tot);
  }

  // zero initial states (capture-safe stream memsets)
  hipMemsetAsync(h1a, 0, h1Elems * 2, stream);
  hipMemsetAsync(c1,  0, h1Elems * 4, stream);
  hipMemsetAsync(h2a, 0, h2Elems * 2, stream);
  hipMemsetAsync(c2,  0, h2Elems * 4, stream);

  // layer 1 recurrence + pooling
  {
    _Float16* hin = h1a;
    _Float16* hout = h1b;
    for (int t = 0; t < kT; ++t) {
      lstm1_step_kernel<<<dim3(8, kH1, kB), 256, 0, stream>>>(
          x, t, W1pk, U1pk, b1, hin, hout, c1);
      pool_step_kernel<<<(int)((h2Elems + 255) / 256), 256, 0, stream>>>(
          hout, hp + (size_t)t * h2Elems);
      _Float16* tmp = hin; hin = hout; hout = tmp;
    }
  }
  // layer 2 recurrence
  {
    _Float16* hin = h2a;
    _Float16* hout = h2b;
    for (int t = 0; t < kT; ++t) {
      lstm2_step_kernel<<<dim3(4, kH2, kB), 256, 0, stream>>>(
          hp + (size_t)t * h2Elems, W2pk, U2pk, b2, hin, hout, c2,
          h2s + (size_t)t * h2Elems);
      _Float16* tmp = hin; hin = hout; hout = tmp;
    }
  }
  // upsample + final conv + residual
  final_conv_kernel<<<dim3(kH1, kB * kT), 256, 0, stream>>>(
      h2s, Wcpk, bc, x, out, res);
}
